// SelfAttention_35545149341704
// MI455X (gfx1250) — compile-verified, matrix operand order checked by reading, not run on hardware
//
#include <hip/hip_runtime.h>

typedef __attribute__((ext_vector_type(16))) __bf16 v16bf;
typedef __attribute__((ext_vector_type(8)))  float  v8f;

#define B_ 16
#define L_ 680
#define C_ 1024
#define H_ 16
#define D_ 64
#define M_ (B_ * L_)          // 10880
#define MAX_SCALE_LOG 4.605170185988091f

// ---------- helpers ----------

// Load a 16x32 bf16 A/B fragment. Per ISA: lane holds row (A) / col (B) = lane&15,
// element e -> k = (e<8 ? e : e+8) + 8*(lane>>4).  rowptr = start of this lane's row.
__device__ __forceinline__ v16bf ld_frag(const __bf16* rowptr, int kbase, int khalf) {
    const __bf16* p0 = rowptr + kbase + (khalf << 3);
    const __bf16* p1 = p0 + 16;
    v16bf f;
#pragma unroll
    for (int i = 0; i < 8; ++i) { f[i] = p0[i]; f[8 + i] = p1[i]; }
    return f;
}

__device__ __forceinline__ v8f wmma_bf16(v16bf a, v16bf b, v8f c) {
    return __builtin_amdgcn_wmma_f32_16x16x32_bf16(false, a, false, b, (short)0, c, false, false);
}

// ---------- fp32 -> bf16 convert ----------
__global__ void cvt_f32_bf16(const float* __restrict__ in, __bf16* __restrict__ out, int n) {
    int i = blockIdx.x * 256 + threadIdx.x;
    if (i < n) out[i] = (__bf16)in[i];
}

// ---------- GEMM1: qkv = x @ Wqkv^T + bias, fused normalize/scale epilogue ----------
// wave tile: 64(M) x 64(N).  N tile == one (s,h) slab (64 = d).
__global__ __launch_bounds__(256) void gemm_qkv(
    const __bf16* __restrict__ xb, const __bf16* __restrict__ wb,
    const float* __restrict__ qbias, const float* __restrict__ vbias,
    const float* __restrict__ slog,
    __bf16* __restrict__ qo, __bf16* __restrict__ ko, __bf16* __restrict__ vto) {
    const int Nt = 48;                       // 3072/64
    int lane = threadIdx.x & 31;
    int wave = threadIdx.x >> 5;
    int gw = blockIdx.x * 8 + wave;
    int mt = gw / Nt, nt = gw % Nt;
    if (mt >= M_ / 64) return;
    int half = lane >> 4, lm = lane & 15;

    const __bf16* ap = xb + (size_t)(mt * 64 + lm) * C_;     // +mi*16*C_ folds to imm
    const __bf16* bp = wb + (size_t)(nt * 64 + lm) * C_;     // +ni*16*C_ folds to imm

    v8f c[4][4] = {};
    for (int k0 = 0; k0 < C_; k0 += 32) {
        v16bf a[4], b[4];
#pragma unroll
        for (int mi = 0; mi < 4; ++mi) a[mi] = ld_frag(ap + (size_t)mi * 16 * C_, k0, half);
#pragma unroll
        for (int ni = 0; ni < 4; ++ni) b[ni] = ld_frag(bp + (size_t)ni * 16 * C_, k0, half);
#pragma unroll
        for (int ni = 0; ni < 4; ++ni)
#pragma unroll
            for (int mi = 0; mi < 4; ++mi)
                c[mi][ni] = wmma_bf16(a[mi], b[ni], c[mi][ni]);
    }

    // epilogue
    int j0 = nt * 64;
    int s = j0 >> 10;               // 0=q 1=k 2=v
    int h = (j0 & 1023) >> 6;
    float bias[4];
#pragma unroll
    for (int ni = 0; ni < 4; ++ni) {
        int col = h * 64 + ni * 16 + lm;
        bias[ni] = (s == 0) ? qbias[col] : ((s == 2) ? vbias[col] : 0.0f);
    }
    float sc = 1.0f;
    if (s == 0) sc = __expf(fminf(slog[h], MAX_SCALE_LOG));

#pragma unroll
    for (int mi = 0; mi < 4; ++mi) {
#pragma unroll
        for (int r = 0; r < 8; ++r) {
            float vv[4];
#pragma unroll
            for (int ni = 0; ni < 4; ++ni) vv[ni] = c[mi][ni][r] + bias[ni];
            int m = mt * 64 + mi * 16 + (half << 3) + r;
            int bb = m / L_, l = m - bb * L_;
            if (s < 2) {
                float ss = vv[0]*vv[0] + vv[1]*vv[1] + vv[2]*vv[2] + vv[3]*vv[3];
                ss += __shfl_xor(ss, 1); ss += __shfl_xor(ss, 2);
                ss += __shfl_xor(ss, 4); ss += __shfl_xor(ss, 8);
                float rn = 1.0f / fmaxf(sqrtf(ss), 1e-12f);
                if (s == 0) rn *= sc;
                __bf16* dst = (s == 0 ? qo : ko) +
                              ((size_t)(bb * H_ + h) * L_ + l) * D_ + lm;
#pragma unroll
                for (int ni = 0; ni < 4; ++ni) dst[ni * 16] = (__bf16)(vv[ni] * rn);
            } else {
                __bf16* dst = vto + ((size_t)(bb * H_ + h) * D_) * L_ + l;
#pragma unroll
                for (int ni = 0; ni < 4; ++ni)
                    dst[(size_t)(ni * 16 + lm) * L_] = (__bf16)vv[ni];
            }
        }
    }
}

// ---------- attention: one wave per (b, h, segment, m-tile) ----------
#define PSTR 272
__global__ __launch_bounds__(32) void attn(
    const __bf16* __restrict__ qb, const __bf16* __restrict__ kb,
    const __bf16* __restrict__ vt, __bf16* __restrict__ ao) {
    __shared__ float  Sf[16][PSTR];
    __shared__ __bf16 Pl[16][PSTR];

    int lane = threadIdx.x & 31;
    int half = lane >> 4, lm = lane & 15;
    int b = blockIdx.z, h = blockIdx.y, t = blockIdx.x;

    const int sstart[10] = {0, 1, 5, 14, 30, 55, 91, 155, 255, 424};
    const int slen[10]   = {1, 4, 9, 16, 25, 36, 64, 100, 169, 256};
    const int tstart[10] = {0, 1, 2, 3, 4, 6, 9, 13, 20, 31};   // cum tiles
    int seg = 0;
#pragma unroll
    for (int i = 1; i < 10; ++i) if (t >= tstart[i]) seg = i;
    int mt = t - tstart[seg];
    int s0 = sstart[seg], ln = slen[seg];
    int nfull = (ln + 15) >> 4;

    const __bf16* qbase = qb + ((size_t)(b * H_ + h) * L_) * D_;
    const __bf16* kbase = kb + ((size_t)(b * H_ + h) * L_) * D_;
    const __bf16* vbase = vt + ((size_t)(b * H_ + h) * D_) * L_;

    int qrow = s0 + mt * 16 + lm;
    v16bf qa0 = ld_frag(qbase + (size_t)qrow * D_, 0, half);
    v16bf qa1 = ld_frag(qbase + (size_t)qrow * D_, 32, half);

    float mx[8];
#pragma unroll
    for (int r = 0; r < 8; ++r) mx[r] = -3.0e38f;

    // pass 1: S = q @ k^T  (stage to LDS)
    for (int nt2 = 0; nt2 < nfull; ++nt2) {
        int krow = s0 + nt2 * 16 + lm;
        v16bf kb0 = ld_frag(kbase + (size_t)krow * D_, 0, half);
        v16bf kb1 = ld_frag(kbase + (size_t)krow * D_, 32, half);
        v8f cacc = {};
        cacc = wmma_bf16(qa0, kb0, cacc);
        cacc = wmma_bf16(qa1, kb1, cacc);
        bool colok = (nt2 * 16 + lm) < ln;
#pragma unroll
        for (int r = 0; r < 8; ++r) {
            float v = colok ? cacc[r] : -3.0e38f;
            Sf[(half << 3) + r][nt2 * 16 + lm] = v;
            mx[r] = fmaxf(mx[r], v);
        }
    }
#pragma unroll
    for (int r = 0; r < 8; ++r) {
        mx[r] = fmaxf(mx[r], __shfl_xor(mx[r], 1));
        mx[r] = fmaxf(mx[r], __shfl_xor(mx[r], 2));
        mx[r] = fmaxf(mx[r], __shfl_xor(mx[r], 4));
        mx[r] = fmaxf(mx[r], __shfl_xor(mx[r], 8));
    }

    // pass 2: P = exp(S - mx), row sums, bf16 to LDS
    float sm[8] = {};
    for (int nt2 = 0; nt2 < nfull; ++nt2) {
#pragma unroll
        for (int r = 0; r < 8; ++r) {
            float p = __expf(Sf[(half << 3) + r][nt2 * 16 + lm] - mx[r]);
            sm[r] += p;
            Pl[(half << 3) + r][nt2 * 16 + lm] = (__bf16)p;
        }
    }
    if (nfull & 1) {
#pragma unroll
        for (int r = 0; r < 8; ++r)
            Pl[(half << 3) + r][nfull * 16 + lm] = (__bf16)0.0f;
    }
#pragma unroll
    for (int r = 0; r < 8; ++r) {
        sm[r] += __shfl_xor(sm[r], 1); sm[r] += __shfl_xor(sm[r], 2);
        sm[r] += __shfl_xor(sm[r], 4); sm[r] += __shfl_xor(sm[r], 8);
    }

    // O = P @ V  (V read from transposed layout: contiguous along K)
    v8f o[4] = {};
    int nk = (nfull + 1) >> 1;
    for (int kt = 0; kt < nk; ++kt) {
        v16bf pa = ld_frag(&Pl[lm][0], kt * 32, half);
        v16bf vb[4];
#pragma unroll
        for (int dn = 0; dn < 4; ++dn)
            vb[dn] = ld_frag(vbase + (size_t)(dn * 16 + lm) * L_, s0 + kt * 32, half);
#pragma unroll
        for (int dn = 0; dn < 4; ++dn) o[dn] = wmma_bf16(pa, vb[dn], o[dn]);
    }

    // normalize + store row-major (B*L, C) bf16
#pragma unroll
    for (int r = 0; r < 8; ++r) {
        int ml = (half << 3) + r;
        if (mt * 16 + ml < ln) {
            int l = s0 + mt * 16 + ml;
            float inv = 1.0f / sm[r];
            size_t off = ((size_t)b * L_ + l) * C_ + h * D_;
#pragma unroll
            for (int dn = 0; dn < 4; ++dn)
                ao[off + dn * 16 + lm] = (__bf16)(o[dn][r] * inv);
        }
    }
}

// ---------- GEMM2: out = att @ Wproj^T + proj_b (fp32 out) ----------
// wave tile: 64(M) x 64(N)
__global__ __launch_bounds__(256) void gemm_proj(
    const __bf16* __restrict__ ab, const __bf16* __restrict__ wb,
    const float* __restrict__ pbias, float* __restrict__ out) {
    const int Nt = 16;                        // 1024/64
    int lane = threadIdx.x & 31;
    int wave = threadIdx.x >> 5;
    int gw = blockIdx.x * 8 + wave;
    int mt = gw / Nt, nt = gw % Nt;
    if (mt >= M_ / 64) return;
    int half = lane >> 4, lm = lane & 15;

    const __bf16* ap = ab + (size_t)(mt * 64 + lm) * C_;
    const __bf16* bp = wb + (size_t)(nt * 64 + lm) * C_;

    v8f c[4][4] = {};
    for (int k0 = 0; k0 < C_; k0 += 32) {
        v16bf a[4], b[4];
#pragma unroll
        for (int mi = 0; mi < 4; ++mi) a[mi] = ld_frag(ap + (size_t)mi * 16 * C_, k0, half);
#pragma unroll
        for (int ni = 0; ni < 4; ++ni) b[ni] = ld_frag(bp + (size_t)ni * 16 * C_, k0, half);
#pragma unroll
        for (int ni = 0; ni < 4; ++ni)
#pragma unroll
            for (int mi = 0; mi < 4; ++mi)
                c[mi][ni] = wmma_bf16(a[mi], b[ni], c[mi][ni]);
    }

    float bias[4];
#pragma unroll
    for (int ni = 0; ni < 4; ++ni) bias[ni] = pbias[nt * 64 + ni * 16 + lm];
#pragma unroll
    for (int mi = 0; mi < 4; ++mi) {
#pragma unroll
        for (int r = 0; r < 8; ++r) {
            int m = mt * 64 + mi * 16 + (half << 3) + r;
#pragma unroll
            for (int ni = 0; ni < 4; ++ni)
                out[(size_t)m * C_ + nt * 64 + ni * 16 + lm] = c[mi][ni][r] + bias[ni];
        }
    }
}

// ---------- launch ----------
extern "C" void kernel_launch(void* const* d_in, const int* in_sizes, int n_in,
                              void* d_out, int out_size, void* d_ws, size_t ws_size,
                              hipStream_t stream) {
    const float* x     = (const float*)d_in[0];
    const float* wqkv  = (const float*)d_in[2];
    const float* qbias = (const float*)d_in[3];
    const float* vbias = (const float*)d_in[4];
    const float* slog  = (const float*)d_in[5];
    const float* wproj = (const float*)d_in[6];
    const float* pbias = (const float*)d_in[7];
    float* out = (float*)d_out;

    char* w = (char*)d_ws;
    const size_t SZ_X  = (size_t)M_ * C_ * 2;        // 22,282,240
    const size_t SZ_WQ = (size_t)3 * C_ * C_ * 2;    //  6,291,456
    const size_t SZ_WP = (size_t)C_ * C_ * 2;        //  2,097,152
    const size_t SZ_Q  = (size_t)B_ * H_ * L_ * D_ * 2;
    __bf16* xb  = (__bf16*)(w);
    __bf16* wqb = (__bf16*)(w + SZ_X);
    __bf16* wpb = (__bf16*)(w + SZ_X + SZ_WQ);
    __bf16* qbf = (__bf16*)(w + SZ_X + SZ_WQ + SZ_WP);
    __bf16* kbf = (__bf16*)(w + SZ_X + SZ_WQ + SZ_WP + SZ_Q);
    __bf16* vtb = (__bf16*)(w + SZ_X + SZ_WQ + SZ_WP + 2 * SZ_Q);
    __bf16* atb = (__bf16*)(w + SZ_X + SZ_WQ + SZ_WP + 3 * SZ_Q);

    int nx = M_ * C_;            // 11,141,120
    int nwq = 3 * C_ * C_;       //  3,145,728
    int nwp = C_ * C_;           //  1,048,576
    cvt_f32_bf16<<<(nx  + 255) / 256, 256, 0, stream>>>(x, xb, nx);
    cvt_f32_bf16<<<(nwq + 255) / 256, 256, 0, stream>>>(wqkv, wqb, nwq);
    cvt_f32_bf16<<<(nwp + 255) / 256, 256, 0, stream>>>(wproj, wpb, nwp);

    // GEMM1: 170 m-tiles x 48 n-tiles = 8160 waves = 1020 blocks
    gemm_qkv<<<1020, 256, 0, stream>>>(xb, wqb, qbias, vbias, slog, qbf, kbf, vtb);

    // attention: 47 m-tiles across segments, per (h, b)
    dim3 ag(47, H_, B_);
    attn<<<ag, 32, 0, stream>>>(qbf, kbf, vtb, atb);

    // GEMM2: 170 x 16 = 2720 waves = 340 blocks
    gemm_proj<<<340, 256, 0, stream>>>(atb, wpb, pbias, out);
}
